// SelfAttention_11854109737180
// MI455X (gfx1250) — matrix operand order repaired and checked by reading, not compiled, not run on hardware
//
#include <hip/hip_runtime.h>

// ---------------- problem constants ----------------
constexpr int B  = 2;
constexpr int C  = 256;
constexpr int N  = 4096;
constexpr int H  = 8;
constexpr int HD = 32;            // head dim == WMMA K for bf16
constexpr int O3 = 3 * C;         // 768 qkv rows
constexpr float SCALE  = 0.17677669529663687f;          // 32^-0.5
constexpr float LOG2E  = 1.44269504088896340736f;
constexpr float QSCALE = SCALE * LOG2E;                 // logits in log2 domain

typedef __attribute__((ext_vector_type(16))) __bf16 v16bf;
typedef __attribute__((ext_vector_type(8)))  __bf16 v8bf;
typedef __attribute__((ext_vector_type(8)))  float  v8f;

__device__ __forceinline__ unsigned short f2bf(float f) {
    unsigned int u = __float_as_uint(f);
    unsigned int r = ((u >> 16) & 1u) + 0x7FFFu;   // round-to-nearest-even
    return (unsigned short)((u + r) >> 16);
}

__device__ __forceinline__ v16bf joinA(v8bf lo, v8bf hi) {
    v16bf a;
#pragma unroll
    for (int i = 0; i < 8; ++i) { a[i] = lo[i]; a[i + 8] = hi[i]; }
    return a;
}

// ---------------- 1) convert x (B,C,N) f32 -> xT (B,N,C) bf16 ----------------
__global__ void sa_convert_x(const float* __restrict__ x, unsigned short* __restrict__ xT) {
    size_t i = (size_t)blockIdx.x * blockDim.x + threadIdx.x;
    if (i >= (size_t)B * C * N) return;
    int n = (int)(i % N);
    size_t t = i / N;
    int c = (int)(t % C);
    int b = (int)(t / C);
    xT[((size_t)b * N + n) * C + c] = f2bf(x[i]);
}

// ---------------- 1b) convert weight matrices f32 -> bf16 (same layout) ------
__global__ void sa_convert_w(const float* __restrict__ src, unsigned short* __restrict__ dst, int count) {
    int i = blockIdx.x * blockDim.x + threadIdx.x;
    if (i < count) dst[i] = f2bf(src[i]);
}

// ---------------- 2) QKV projection: qkv[o,n] = W[o,:] . x[:,n] + b[o] -------
// one wave -> one 16x16 (o,n) tile, K-loop over C in steps of 32 (8 WMMAs)
__global__ void __launch_bounds__(256)
sa_qkv_gemm(const unsigned short* __restrict__ Wq,   // [O3][C] bf16
            const unsigned short* __restrict__ xT,   // [B][N][C] bf16
            const float* __restrict__ bias,          // [O3]
            unsigned short* __restrict__ Q,          // [B][H][N][HD] (log2-domain pre-scaled)
            unsigned short* __restrict__ Kmat,       // [B][H][N][HD]
            unsigned short* __restrict__ Vt)         // [B][H][HD][N]
{
    const int lane = threadIdx.x & 31;
    const int wv   = threadIdx.x >> 5;
    const int b    = blockIdx.x / (O3 / 16);
    const int o0   = (blockIdx.x % (O3 / 16)) * 16;
    const int n0   = (blockIdx.y * 8 + wv) * 16;
    const int half = lane >> 4;
    const int l16  = lane & 15;

    v8f acc = {0.f, 0.f, 0.f, 0.f, 0.f, 0.f, 0.f, 0.f};
    const unsigned short* wrow = Wq + (size_t)(o0 + l16) * C;            // A row m
    const unsigned short* xrow = xT + ((size_t)b * N + (n0 + l16)) * C;  // B col n
#pragma unroll
    for (int c0 = 0; c0 < C; c0 += 32) {
        v16bf a  = *(const v16bf*)(wrow + c0 + half * 16);
        v16bf bm = *(const v16bf*)(xrow + c0 + half * 16);
        acc = __builtin_amdgcn_wmma_f32_16x16x32_bf16(false, a, false, bm,
                                                      (short)0, acc, false, false);
    }
    const int which = o0 >> 8;        // 0=q 1=k 2=v
    const int h     = (o0 >> 5) & 7;
    const int d0    = o0 & 31;        // 0 or 16
    const int n     = n0 + l16;
#pragma unroll
    for (int r = 0; r < 8; ++r) {
        const int mrow = half * 8 + r;
        const int o    = o0 + mrow;
        const int d    = d0 + mrow;
        const float v  = acc[r] + bias[o];
        if (which == 0)
            Q[((size_t)(b * H + h) * N + n) * HD + d] = f2bf(v * QSCALE);
        else if (which == 1)
            Kmat[((size_t)(b * H + h) * N + n) * HD + d] = f2bf(v);
        else
            Vt[((size_t)(b * H + h) * HD + d) * N + n] = f2bf(v);
    }
}

// ---------------- 3) fused flash attention -----------------------------------
// block = 8 waves; wave -> 16 query rows; keys streamed in chunks of 64:
//   4 QK^T WMMAs -> single cross-lane max reduction -> exp2 (log2-domain logits)
//   -> P staged in LDS -> 4 PV WMMAs + 2 row-sum WMMAs (ones-column B matrix).
__global__ void __launch_bounds__(256)
sa_flash_attn(const unsigned short* __restrict__ Q,     // [B][H][N][HD]
              const unsigned short* __restrict__ Kmat,  // [B][H][N][HD]
              const unsigned short* __restrict__ Vt,    // [B][H][HD][N]
              unsigned short* __restrict__ OT)          // [B][N][C] bf16
{
    __shared__ __align__(64) unsigned short Plds[8][16][64];  // per-wave P staging (16 KB)
    const int lane = threadIdx.x & 31;
    const int wv   = threadIdx.x >> 5;
    const int bh   = blockIdx.x;              // b*H + h
    const int b    = bh >> 3;
    const int h    = bh & 7;
    const int q0   = blockIdx.y * 128 + wv * 16;
    const int half = lane >> 4;
    const int l16  = lane & 15;

    // Q A-tile (16 queries x HD=32), loaded once
    const unsigned short* qrow = Q + ((size_t)bh * N + (q0 + l16)) * HD;
    v16bf qa = *(const v16bf*)(qrow + half * 16);

    // B-matrix with ones in column 0: turns row-sum of P into a WMMA
    v16bf onesb;
    {
        const __bf16 one = (__bf16)1.0f, nil = (__bf16)0.0f;
#pragma unroll
        for (int i = 0; i < 16; ++i) onesb[i] = (l16 == 0) ? one : nil;
    }

    float mi[8];
    v8f acc0 = {0.f, 0.f, 0.f, 0.f, 0.f, 0.f, 0.f, 0.f};  // O columns d0..d15
    v8f acc1 = acc0;                                        // O columns d16..d31
    v8f acc2 = acc0;                                        // softmax denominator (col 0)
#pragma unroll
    for (int r = 0; r < 8; ++r) mi[r] = -1e30f;

    const unsigned short* kbase = Kmat + (size_t)bh * N * HD;
    const unsigned short* vbase = Vt + (size_t)bh * HD * N;

    for (int kb = 0; kb < N; kb += 64) {
        // 4 score tiles (16 queries x 64 keys), logits already in log2 domain
        const v8f zero = {0.f, 0.f, 0.f, 0.f, 0.f, 0.f, 0.f, 0.f};
        v8f s[4];
#pragma unroll
        for (int t = 0; t < 4; ++t) {
            v16bf kt = *(const v16bf*)(kbase + (size_t)(kb + t * 16 + l16) * HD + half * 16);
            s[t] = __builtin_amdgcn_wmma_f32_16x16x32_bf16(false, qa, false, kt,
                                                           (short)0, zero, false, false);
        }
        // online softmax: one max reduction per 64 keys, sums via WMMA below
#pragma unroll
        for (int r = 0; r < 8; ++r) {
            float vmax = fmaxf(fmaxf(s[0][r], s[1][r]), fmaxf(s[2][r], s[3][r]));
            vmax = fmaxf(vmax, __shfl_xor(vmax, 1, 32));
            vmax = fmaxf(vmax, __shfl_xor(vmax, 2, 32));
            vmax = fmaxf(vmax, __shfl_xor(vmax, 4, 32));
            vmax = fmaxf(vmax, __shfl_xor(vmax, 8, 32));
            const float mn = fmaxf(mi[r], vmax);
            const float al = exp2f(mi[r] - mn);
            mi[r] = mn;
            acc0[r] *= al;
            acc1[r] *= al;
            acc2[r] *= al;
            const int mrow = half * 8 + r;
#pragma unroll
            for (int t = 0; t < 4; ++t)
                Plds[wv][mrow][t * 16 + l16] = f2bf(exp2f(s[t][r] - mn));
        }
        // in-wave LDS RAW: stores above feed A-layout reads below
        asm volatile("s_wait_dscnt 0x0" ::: "memory");
        const unsigned short* prow = &Plds[wv][l16][0];
#pragma unroll
        for (int sc = 0; sc < 2; ++sc) {     // two 32-key sub-chunks
            v16bf pa = *(const v16bf*)(prow + sc * 32 + half * 16);
            v16bf vb0 = *(const v16bf*)(vbase + (size_t)l16 * N        + kb + sc * 32 + half * 16);
            v16bf vb1 = *(const v16bf*)(vbase + (size_t)(16 + l16) * N + kb + sc * 32 + half * 16);
            acc0 = __builtin_amdgcn_wmma_f32_16x16x32_bf16(false, pa, false, vb0,
                                                           (short)0, acc0, false, false);
            acc1 = __builtin_amdgcn_wmma_f32_16x16x32_bf16(false, pa, false, vb1,
                                                           (short)0, acc1, false, false);
            acc2 = __builtin_amdgcn_wmma_f32_16x16x32_bf16(false, pa, false, onesb,
                                                           (short)0, acc2, false, false);
        }
    }
    // normalize (denominator lives in column 0 of acc2) and store O^T
#pragma unroll
    for (int r = 0; r < 8; ++r) {
        const int   mrow = half * 8 + r;
        const float li   = __shfl(acc2[r], half * 16, 32);   // broadcast col 0 of this half
        const float inv  = 1.0f / li;
        const size_t row = ((size_t)b * N + (q0 + mrow)) * C;
        OT[row + h * HD + l16]      = f2bf(acc0[r] * inv);
        OT[row + h * HD + 16 + l16] = f2bf(acc1[r] * inv);
    }
}

// ---------------- 4) output projection: y = Wo @ O + b (fp32 out) ------------
__global__ void __launch_bounds__(256)
sa_out_gemm(const unsigned short* __restrict__ Wo,   // [C][C] bf16
            const unsigned short* __restrict__ OT,   // [B][N][C] bf16
            const float* __restrict__ bias,          // [C]
            float* __restrict__ y)                   // [B][C][N] f32
{
    const int lane = threadIdx.x & 31;
    const int wv   = threadIdx.x >> 5;
    const int b    = blockIdx.x >> 4;           // C/16 = 16 o-tiles
    const int o0   = (blockIdx.x & 15) * 16;
    const int n0   = (blockIdx.y * 8 + wv) * 16;
    const int half = lane >> 4;
    const int l16  = lane & 15;

    v8f acc = {0.f, 0.f, 0.f, 0.f, 0.f, 0.f, 0.f, 0.f};
    const unsigned short* wrow = Wo + (size_t)(o0 + l16) * C;
    const unsigned short* xrow = OT + ((size_t)b * N + (n0 + l16)) * C;
#pragma unroll
    for (int c0 = 0; c0 < C; c0 += 32) {
        v16bf a  = *(const v16bf*)(wrow + c0 + half * 16);
        v16bf bm = *(const v16bf*)(xrow + c0 + half * 16);
        acc = __builtin_amdgcn_wmma_f32_16x16x32_bf16(false, a, false, bm,
                                                      (short)0, acc, false, false);
    }
#pragma unroll
    for (int r = 0; r < 8; ++r) {
        const int o = o0 + half * 8 + r;
        y[((size_t)b * C + o) * N + n0 + l16] = acc[r] + bias[o];
    }
}

// ---------------- host side ---------------------------------------------------
extern "C" void kernel_launch(void* const* d_in, const int* in_sizes, int n_in,
                              void* d_out, int out_size, void* d_ws, size_t ws_size,
                              hipStream_t stream) {
    (void)in_sizes; (void)n_in; (void)out_size; (void)ws_size;
    const float* x     = (const float*)d_in[0];
    const float* qkv_w = (const float*)d_in[1];
    const float* qkv_b = (const float*)d_in[2];
    const float* out_w = (const float*)d_in[3];
    const float* out_b = (const float*)d_in[4];
    float* y = (float*)d_out;

    // workspace carve-up (all sizes multiples of 256B)
    char* ws = (char*)d_ws;
    constexpr size_t SZ_XT = (size_t)B * N * C * 2;        // 4 MiB
    constexpr size_t SZ_WQ = (size_t)O3 * C * 2;           // 384 KiB
    constexpr size_t SZ_WO = (size_t)C * C * 2;            // 128 KiB
    constexpr size_t SZ_Q  = (size_t)B * H * N * HD * 2;   // 4 MiB
    unsigned short* xT = (unsigned short*)(ws);
    unsigned short* wq = (unsigned short*)(ws + SZ_XT);
    unsigned short* wo = (unsigned short*)(ws + SZ_XT + SZ_WQ);
    unsigned short* Qb = (unsigned short*)(ws + SZ_XT + SZ_WQ + SZ_WO);
    unsigned short* Kb = (unsigned short*)(ws + SZ_XT + SZ_WQ + SZ_WO + SZ_Q);
    unsigned short* Vt = (unsigned short*)(ws + SZ_XT + SZ_WQ + SZ_WO + 2 * SZ_Q);
    unsigned short* OT = (unsigned short*)(ws + SZ_XT + SZ_WQ + SZ_WO + 3 * SZ_Q);

    // 1) bf16 conversion / layout transforms
    sa_convert_x<<<(B * C * N + 255) / 256, 256, 0, stream>>>(x, xT);
    sa_convert_w<<<(O3 * C + 255) / 256, 256, 0, stream>>>(qkv_w, wq, O3 * C);
    sa_convert_w<<<(C * C + 255) / 256, 256, 0, stream>>>(out_w, wo, C * C);

    // 2) QKV projection (WMMA)
    sa_qkv_gemm<<<dim3(B * (O3 / 16), N / (16 * 8)), 256, 0, stream>>>(
        wq, xT, qkv_b, Qb, Kb, Vt);

    // 3) fused flash attention (WMMA QK^T + online softmax + WMMA PV/row-sums)
    sa_flash_attn<<<dim3(B * H, N / 128), 256, 0, stream>>>(Qb, Kb, Vt, OT);

    // 4) output projection (WMMA) -> fp32
    sa_out_gemm<<<dim3(B * (C / 16), N / (16 * 8)), 256, 0, stream>>>(
        wo, OT, out_b, y);
}
